// AttentionLayer_16879221473394
// MI455X (gfx1250) — matrix operand order repaired and checked by reading, not compiled, not run on hardware
//
#include <hip/hip_runtime.h>
#include <cstdint>
#include <cstddef>

typedef __attribute__((ext_vector_type(16))) _Float16 v16h;
typedef __attribute__((ext_vector_type(8)))  float    v8f;

#define B_   64
#define TX_  2048
#define DA_  256
#define DS_  256
#define H_   50
#define HP_  64          // H padded to multiple of 16
#define NT_  4           // HP_/16 N-tiles
#define KB_  8           // DA_/32 K-steps of 32

// ---------------------------------------------------------------------------
// Kernel 0a: pack W1[:256, :50] (fp32 row-major [k][h]) into f16 WMMA B layout.
// Layout: w1p[ ((nt*KB_ + kb)*32 + lane)*16 + e ]
//   n = nt*16 + (lane&15); hi = lane>>4; K = kb*32 + hi*8 + e + (e>=8 ? 8 : 0)
// (16-bit B 32x16: elems 0..7 hold K = hi*8 + e, elems 8..15 hold
//  K = 16 + hi*8 + (e-8), matching the A fragment order.)
// ---------------------------------------------------------------------------
__global__ void pack_w1(const float* __restrict__ W1, _Float16* __restrict__ w1p) {
    int idx = blockIdx.x * blockDim.x + threadIdx.x;     // 0 .. 16383
    if (idx >= NT_ * KB_ * 32 * 16) return;
    int e    = idx & 15;
    int lane = (idx >> 4) & 31;
    int kb   = (idx >> 9) & (KB_ - 1);
    int nt   = idx >> 12;
    int n    = nt * 16 + (lane & 15);
    int hi   = lane >> 4;
    int k    = kb * 32 + hi * 8 + e + ((e >> 3) << 3);   // k < 256
    float v  = (n < H_) ? W1[k * H_ + n] : 0.0f;
    w1p[idx] = (_Float16)v;
}

// ---------------------------------------------------------------------------
// Kernel 0b: cs[b][n] = b1[n] + sum_k s[b,k] * W1[256+k, n]   (n<50, else 0)
// ---------------------------------------------------------------------------
__global__ void comp_cs(const float* __restrict__ s, const float* __restrict__ W1,
                        const float* __restrict__ b1, float* __restrict__ cs) {
    int b = blockIdx.x;
    int n = threadIdx.x;                                 // 64 threads
    float acc = 0.0f;
    if (n < H_) {
        acc = b1[n];
        const float* sb = s + (size_t)b * DS_;
        for (int k = 0; k < DS_; ++k)
            acc = fmaf(sb[k], W1[(size_t)(DA_ + k) * H_ + n], acc);
    }
    cs[b * HP_ + n] = acc;
}

// ---------------------------------------------------------------------------
// Kernel 1: e[b,t] = tanh( W2 . relu( a[b,t,:] @ W1[:256] + cs[b] ) + b2 )
// 4 waves/block, each wave computes a 16(time) x 64(h) tile via 8x4 WMMAs.
// ---------------------------------------------------------------------------
__global__ void __launch_bounds__(128)
fused_e(const float* __restrict__ a, const _Float16* __restrict__ w1p,
        const float* __restrict__ cs, const float* __restrict__ W2,
        const float* __restrict__ b2, float* __restrict__ e_out) {
    const int b     = blockIdx.y;
    const int wave  = threadIdx.x >> 5;
    const int lane  = threadIdx.x & 31;
    const int tbase = blockIdx.x * 64 + wave * 16;
    const int m     = lane & 15;
    const int hi    = lane >> 4;

    // A-frag base: row t = tbase+m; 16-bit A 16x32 layout:
    //  elems 0..7  -> K = kb*32 + hi*8 + [0..8)
    //  elems 8..15 -> K = kb*32 + 16 + hi*8 + [0..8)
    const float* pa = a + ((size_t)b * TX_ + (tbase + m)) * DA_ + hi * 8;
    const v16h*  wb = (const v16h*)w1p;

    v8f acc[NT_] = {};

    #pragma unroll
    for (int kb = 0; kb < KB_; ++kb) {
        const float4* p = (const float4*)(pa + kb * 32);
        float4 f0 = p[0], f1 = p[1];     // K = kb*32 + hi*8 + 0..7
        float4 f2 = p[4], f3 = p[5];     // K = kb*32 + 16 + hi*8 + 0..7
        v16h af;
        af[0]  = (_Float16)f0.x;  af[1]  = (_Float16)f0.y;
        af[2]  = (_Float16)f0.z;  af[3]  = (_Float16)f0.w;
        af[4]  = (_Float16)f1.x;  af[5]  = (_Float16)f1.y;
        af[6]  = (_Float16)f1.z;  af[7]  = (_Float16)f1.w;
        af[8]  = (_Float16)f2.x;  af[9]  = (_Float16)f2.y;
        af[10] = (_Float16)f2.z;  af[11] = (_Float16)f2.w;
        af[12] = (_Float16)f3.x;  af[13] = (_Float16)f3.y;
        af[14] = (_Float16)f3.z;  af[15] = (_Float16)f3.w;

        #pragma unroll
        for (int nt = 0; nt < NT_; ++nt) {
            v16h bf = wb[(nt * KB_ + kb) * 32 + lane];   // contiguous 32B/lane
            acc[nt] = __builtin_amdgcn_wmma_f32_16x16x32_f16(
                false, af, false, bf, (short)0, acc[nt], false, false);
        }
    }

    // Epilogue: relu(acc + cs), weight by W2 (0-padded), reduce over n.
    // acc[nt][r] holds h[t = tbase + r + 8*hi][n = nt*16 + m].
    float partial[8];
    #pragma unroll
    for (int r = 0; r < 8; ++r) partial[r] = 0.0f;

    #pragma unroll
    for (int nt = 0; nt < NT_; ++nt) {
        int   n   = nt * 16 + m;
        float w2v = (n < H_) ? W2[n] : 0.0f;
        float csv = cs[b * HP_ + n];
        #pragma unroll
        for (int r = 0; r < 8; ++r) {
            float h = acc[nt][r] + csv;
            h = fmaxf(h, 0.0f);
            partial[r] = fmaf(w2v, h, partial[r]);
        }
    }

    // Butterfly-reduce over the 16 lanes of each half-wave (wave32).
    #pragma unroll
    for (int r = 0; r < 8; ++r) {
        float v = partial[r];
        v += __shfl_xor(v, 1, 32);
        v += __shfl_xor(v, 2, 32);
        v += __shfl_xor(v, 4, 32);
        v += __shfl_xor(v, 8, 32);
        partial[r] = v;
    }

    if (m == 0) {
        const float bb2 = b2[0];
        const int   t0  = tbase + hi * 8;
        #pragma unroll
        for (int r = 0; r < 8; ++r)
            e_out[(size_t)b * TX_ + t0 + r] = tanhf(partial[r] + bb2);
    }
}

// ---------------------------------------------------------------------------
// Kernel 2: softmax over time per batch.
// ---------------------------------------------------------------------------
__global__ void __launch_bounds__(256)
softmax_t(const float* __restrict__ e, float* __restrict__ alpha) {
    __shared__ float red[256];
    const int b   = blockIdx.x;
    const int tid = threadIdx.x;
    const float* eb = e + (size_t)b * TX_;

    float mx = -3.402823466e+38f;
    for (int t = tid; t < TX_; t += 256) mx = fmaxf(mx, eb[t]);
    red[tid] = mx; __syncthreads();
    for (int s2 = 128; s2 > 0; s2 >>= 1) {
        if (tid < s2) red[tid] = fmaxf(red[tid], red[tid + s2]);
        __syncthreads();
    }
    mx = red[0]; __syncthreads();

    float sum = 0.0f;
    for (int t = tid; t < TX_; t += 256) sum += __expf(eb[t] - mx);
    red[tid] = sum; __syncthreads();
    for (int s2 = 128; s2 > 0; s2 >>= 1) {
        if (tid < s2) red[tid] += red[tid + s2];
        __syncthreads();
    }
    const float inv = 1.0f / red[0];

    for (int t = tid; t < TX_; t += 256)
        alpha[(size_t)b * TX_ + t] = __expf(eb[t] - mx) * inv;
}

// ---------------------------------------------------------------------------
// Kernel 3a: partial context over 256-t slices (deterministic, no atomics).
// ---------------------------------------------------------------------------
__global__ void __launch_bounds__(256)
ctx_partial(const float* __restrict__ a, const float* __restrict__ alpha,
            float* __restrict__ part) {
    const int b     = blockIdx.y;
    const int slice = blockIdx.x;            // 0..7
    const int d     = threadIdx.x;           // 0..255
    const float* ab = a + ((size_t)b * TX_ + slice * 256) * DA_ + d;
    const float* al = alpha + (size_t)b * TX_ + slice * 256;
    float acc = 0.0f;
    for (int t = 0; t < 256; ++t)
        acc = fmaf(al[t], ab[(size_t)t * DA_], acc);
    part[((size_t)b * 8 + slice) * DA_ + d] = acc;
}

// ---------------------------------------------------------------------------
// Kernel 3b: reduce the 8 slices -> context[b, 0, d].
// ---------------------------------------------------------------------------
__global__ void __launch_bounds__(256)
ctx_reduce(const float* __restrict__ part, float* __restrict__ out) {
    int idx = blockIdx.x * blockDim.x + threadIdx.x;   // 0..16383
    int b = idx >> 8, d = idx & 255;
    float acc = 0.0f;
    #pragma unroll
    for (int sl = 0; sl < 8; ++sl)
        acc += part[((size_t)b * 8 + sl) * DA_ + d];
    out[idx] = acc;
}

// ---------------------------------------------------------------------------
extern "C" void kernel_launch(void* const* d_in, const int* in_sizes, int n_in,
                              void* d_out, int out_size, void* d_ws, size_t ws_size,
                              hipStream_t stream) {
    const float* a  = (const float*)d_in[0];   // [64,2048,256]
    const float* s  = (const float*)d_in[1];   // [64,256]
    const float* W1 = (const float*)d_in[2];   // [512,50]
    const float* b1 = (const float*)d_in[3];   // [50]
    const float* W2 = (const float*)d_in[4];   // [50,1]
    const float* b2 = (const float*)d_in[5];   // [1]
    float* out = (float*)d_out;                // [64,1,256]

    char* ws = (char*)d_ws;
    // Workspace layout (256B-aligned regions):
    _Float16* w1p   = (_Float16*)(ws);                               // 32 KB
    float*    cs    = (float*)(ws + 32768);                          // 16 KB
    float*    e     = (float*)(ws + 49152);                          // 512 KB
    float*    alpha = (float*)(ws + 49152 + 524288);                 // 512 KB
    float*    part  = (float*)(ws + 49152 + 2 * 524288);             // 512 KB

    pack_w1   <<<64, 256, 0, stream>>>(W1, w1p);
    comp_cs   <<<B_, 64, 0, stream>>>(s, W1, b1, cs);
    fused_e   <<<dim3(TX_ / 64, B_), 128, 0, stream>>>(a, w1p, cs, W2, b2, e);
    softmax_t <<<B_, 256, 0, stream>>>(e, alpha);
    ctx_partial<<<dim3(8, B_), 256, 0, stream>>>(a, alpha, part);
    ctx_reduce<<<64, 256, 0, stream>>>(part, out);
}
